// Q_network_12171937317191
// MI455X (gfx1250) — matrix operand order repaired and checked by reading, not compiled, not run on hardware
//
#include <hip/hip_runtime.h>
#include <hip/hip_bf16.h>

typedef _Float16 half8 __attribute__((ext_vector_type(8)));
typedef _Float16 v16h  __attribute__((ext_vector_type(16)));
typedef float    v8f   __attribute__((ext_vector_type(8)));

union FragU { v16h v; half8 h[2]; };

// Branch-free tanh: 1 - 2/(exp(2x)+1).  v_exp_f32 + v_rcp_f32, saturates to +/-1
// (rcp(inf)=0 for x->+inf, rcp(1)=1 for x->-inf).  No EXEC-mask branching.
__device__ static __forceinline__ float fast_tanh(float x) {
    float t = __expf(2.0f * x);
    float r = __builtin_amdgcn_rcpf(t + 1.0f);
    return 1.0f - 2.0f * r;
}

// ---- WMMA fragment loaders (contiguous k order, the same in A and B) ----

// A-matrix 16x32 f16: lane&15 = M row; halves 0..15 -> K = kb*32 + (lane>=16?16:0)+h.
// Two contiguous 16-byte LDS reads per lane.
__device__ static __forceinline__ v16h load_fragA(const _Float16* row, int kb, int lane) {
    int koff = kb * 32 + ((lane & 16) ? 16 : 0);
    FragU f;
    f.h[0] = *(const half8*)(row + koff);
    f.h[1] = *(const half8*)(row + koff + 8);
    return f.v;
}

// B-matrix 32x16 f16 pre-packed fragment: 32 lanes x 16 halves, lane-major.
__device__ static __forceinline__ v16h load_fragB_packed(const _Float16* frag, int lane) {
    const half8* p = (const half8*)(frag + lane * 16);
    FragU f;
    f.h[0] = p[0];
    f.h[1] = p[1];
    return f.v;
}

// B-matrix 32x16 built from a ROW-MAJOR [rows x stride] f16 LDS buffer where we
// need B[k][n] = buf[n][k] (i.e. multiplying by buf^T): lane&15 = n (row of buf),
// halves = 16 contiguous k at kb*32 + (lane>=16?16:0).
__device__ static __forceinline__ v16h load_fragB_rowmajor(const _Float16* buf, int stride,
                                                           int nt, int kb, int lane) {
    int n  = nt * 16 + (lane & 15);
    int k0 = kb * 32 + ((lane & 16) ? 16 : 0);
    const _Float16* p = buf + n * stride + k0;
    FragU f;
    f.h[0] = *(const half8*)(p);
    f.h[1] = *(const half8*)(p + 8);
    return f.v;
}

// ---- Dense layer: [32 x K] (LDS f16) @ [K x 16*Ntiles] (packed f16 global) ----
// 8 waves split the 2*Ntiles 16x16 output tiles; f32 accumulate; optional tanh;
// writes f16 (next layer input) or f32 (LayerNorm staging).  B fragments are
// double-buffered so global loads overlap the WMMA chain.
__device__ static __forceinline__ void mlp_layer(
    const _Float16* __restrict__ src, int srcStride, int Kp,
    const _Float16* __restrict__ wp, const float* __restrict__ bias,
    _Float16* dstH, float* dstF, int dstStride, int Ntiles,
    int tid, bool doTanh, bool toF32)
{
    const int wave = tid >> 5, lane = tid & 31;
    __builtin_prefetch(wp, 0, 1);                 // global_prefetch_b8
    const int nTiles = 2 * Ntiles;
    for (int t = wave; t < nTiles; t += 8) {      // wave-uniform: EXEC all ones
        const int mt = t & 1, nt = t >> 1;
        const _Float16* arow  = src + (mt * 16 + (lane & 15)) * srcStride;
        const _Float16* wbase = wp + ((size_t)nt * Kp) * 512;
        const int n  = nt * 16 + (lane & 15);
        const float bv = bias[n];                 // overlaps with WMMA chain below
        v8f acc = {};
        v16h bcur = load_fragB_packed(wbase, lane);
        for (int kb = 0; kb < Kp; ++kb) {
            v16h bnext = bcur;
            if (kb + 1 < Kp) bnext = load_fragB_packed(wbase + (kb + 1) * 512, lane);
            v16h a = load_fragA(arow, kb, lane);
            acc = __builtin_amdgcn_wmma_f32_16x16x32_f16(false, a, false, bcur,
                                                         (short)0, acc, false, false);
            bcur = bnext;
        }
        const int mbase = mt * 16 + ((lane & 16) ? 8 : 0);
#pragma unroll
        for (int r = 0; r < 8; ++r) {
            float v = acc[r] + bv;
            if (doTanh) v = fast_tanh(v);
            if (toF32) dstF[(mbase + r) * dstStride + n] = v;
            else       dstH[(mbase + r) * dstStride + n] = (_Float16)v;
        }
    }
}

// ---- LayerNorm over 256 features, rows split across 8 waves, wave32 shuffle reduce ----
__device__ static __forceinline__ void layer_norm(
    const float* __restrict__ x, const float* __restrict__ w, const float* __restrict__ bgl,
    _Float16* dst, int dstStride, int colOff, int tid)
{
    const int wave = tid >> 5, lane = tid & 31;
    for (int row = wave; row < 32; row += 8) {
        const float* xr = x + row * 256;
        float s = 0.0f;
        for (int c = lane; c < 256; c += 32) s += xr[c];
        for (int off = 16; off; off >>= 1) s += __shfl_xor(s, off, 32);
        const float mu = s * (1.0f / 256.0f);
        float v = 0.0f;
        for (int c = lane; c < 256; c += 32) { float d = xr[c] - mu; v += d * d; }
        for (int off = 16; off; off >>= 1) v += __shfl_xor(v, off, 32);
        const float rstd = rsqrtf(v * (1.0f / 256.0f) + 1e-5f);
        for (int c = lane; c < 256; c += 32) {
            float y = (xr[c] - mu) * rstd * w[c] + bgl[c];
            dst[row * dstStride + colOff + c] = (_Float16)y;
        }
    }
}

// ---- Weight packing: f32 row-major [K x N] -> f16 WMMA B fragments, zero-padded K ----
__global__ void pack_weight_kernel(const float* __restrict__ W, _Float16* __restrict__ Pk,
                                   int K, int Ncols)
{
    const int kb = blockIdx.x, nt = blockIdx.y, lane = threadIdx.x;
    const int Kp = gridDim.x;
    const int n = nt * 16 + (lane & 15);
    const int kbase = kb * 32 + ((lane & 16) ? 16 : 0);
    _Float16* dst = Pk + (((size_t)nt * Kp + kb) * 32 + lane) * 16;
#pragma unroll
    for (int h = 0; h < 16; ++h) {
        int k = kbase + h;
        float v = (k < K) ? W[(size_t)k * Ncols + n] : 0.0f;
        dst[h] = (_Float16)v;
    }
}

struct QNetParams {
    const float *states, *policies, *actions;
    const float *se_b1, *se_b2, *k_b1, *k_b2, *q_b1, *q_b2;
    const float *sa_b1, *sa_b2, *av_b1, *av_b2, *ca_b1, *ca_b2;
    const float *f_b1, *f_b2, *f_b3;
    const float *ln1_w, *ln1_b, *ln2_w, *ln2_b;
    const _Float16 *PseW1, *PseW2, *PkW1, *PkW2, *PqW1, *PqW2;
    const _Float16 *PsaW1, *PsaW2, *PavW1, *PavW2, *PcaW1, *PcaW2;
    const _Float16 *PfW1, *PfW2, *PfW3;
    float* out;
};

// One workgroup per batch element b: all 32 nodes' MLPs + attention fused.
__global__ __launch_bounds__(256) void qnet_fused_kernel(QNetParams P)
{
    extern __shared__ char smem[];
    _Float16* XA   = (_Float16*)smem;        // 32 x 160  [state|action|0pad] f16
    _Float16* E    = XA   + 32 * 160;        // 32 x 256  sq_embed
    _Float16* Kf   = E    + 32 * 256;        // 32 x 256  key
    _Float16* Qf   = Kf   + 32 * 256;        // 32 x 256  query
    _Float16* OA   = Qf   + 32 * 256;        // 32 x 256  oa_embed
    _Float16* AV   = OA   + 32 * 256;        // 32 x 256  att_vals
    _Float16* FEAT = AV   + 32 * 256;        // 32 x 512  [cur | node_feat] post-LN
    _Float16* HID  = FEAT + 32 * 512;        // 32 x 256  MLP hidden scratch
    _Float16* WtH  = HID  + 32 * 256;        // 32 x 32   softmax weights f16 (diag 0)
    float* TMP32 = (float*)(WtH + 32 * 32);  // 32 x 256  pre-LN f32
    float* S32   = TMP32 + 32 * 256;         // 32 x 32   scores
    float* Qb    = S32   + 32 * 32;          // 32 x 16   Q logits f32
    float* SumE  = Qb    + 32 * 16;          // 256 colsum of E
    float* SumOA = SumE  + 256;              // 256 colsum of OA

    const int tid = threadIdx.x;
    const int b = blockIdx.x;
    const int lane = tid & 31;
    const int wave = tid >> 5;

    // 1) stage states+actions (f32 global -> f16 LDS), zero-pad cols 144..159
    const float* st = P.states  + (size_t)b * 32 * 128;
    const float* ac = P.actions + (size_t)b * 32 * 16;
    for (int idx = tid; idx < 32 * 160; idx += 256) {
        int m = idx / 160, c = idx - m * 160;
        float v = (c < 128) ? st[m * 128 + c] : (c < 144 ? ac[m * 16 + (c - 128)] : 0.0f);
        XA[idx] = (_Float16)v;
    }
    __syncthreads();

    // 2) sq_embed E = se-MLP(states)   (sk_embed is a gather of E -> never recomputed)
    mlp_layer(XA, 160, 4, P.PseW1, P.se_b1, HID, nullptr, 256, 16, tid, true, false);
    __syncthreads();
    mlp_layer(HID, 256, 8, P.PseW2, P.se_b2, E, nullptr, 256, 16, tid, true, false);
    __syncthreads();
    { float s = 0.0f; for (int i = 0; i < 32; ++i) s += (float)E[i * 256 + tid]; SumE[tid] = s; }
    __syncthreads();

    // 3) key / 4) query (per node; key_obs is a gather of Kf)
    mlp_layer(E, 256, 8, P.PkW1, P.k_b1, HID, nullptr, 256, 16, tid, true, false); __syncthreads();
    mlp_layer(HID, 256, 8, P.PkW2, P.k_b2, Kf, nullptr, 256, 16, tid, true, false); __syncthreads();
    mlp_layer(E, 256, 8, P.PqW1, P.q_b1, HID, nullptr, 256, 16, tid, true, false); __syncthreads();
    mlp_layer(HID, 256, 8, P.PqW2, P.q_b2, Qf, nullptr, 256, 16, tid, true, false); __syncthreads();

    // 5) oa_embed per node (K padded 144->160)
    mlp_layer(XA, 160, 5, P.PsaW1, P.sa_b1, HID, nullptr, 256, 16, tid, true, false); __syncthreads();
    mlp_layer(HID, 256, 8, P.PsaW2, P.sa_b2, OA, nullptr, 256, 16, tid, true, false); __syncthreads();
    { float s = 0.0f; for (int i = 0; i < 32; ++i) s += (float)OA[i * 256 + tid]; SumOA[tid] = s; }
    __syncthreads();

    // 6) att_vals per node
    mlp_layer(OA, 256, 8, P.PavW1, P.av_b1, HID, nullptr, 256, 16, tid, true, false); __syncthreads();
    mlp_layer(HID, 256, 8, P.PavW2, P.av_b2, AV, nullptr, 256, 16, tid, true, false); __syncthreads();

    // 7) cur = LN2( ca-MLP(states) + (SumE-E_i)/31 + E_i ) -> FEAT[:, 0:256]
    mlp_layer(XA, 160, 4, P.PcaW1, P.ca_b1, HID, nullptr, 256, 16, tid, true, false); __syncthreads();
    mlp_layer(HID, 256, 8, P.PcaW2, P.ca_b2, nullptr, TMP32, 256, 16, tid, true, true); __syncthreads();
    for (int idx = tid; idx < 32 * 256; idx += 256) {
        int c = idx & 255;
        float e = (float)E[idx];
        TMP32[idx] += e + (SumE[c] - e) * (1.0f / 31.0f);
    }
    __syncthreads();
    layer_norm(TMP32, P.ln2_w, P.ln2_b, FEAT, 512, 0, tid);
    __syncthreads();

    // 8) scores S = Qf @ Kf^T / sqrt(256); 4 tiles on waves 0..3
    for (int t = wave; t < 4; t += 8) {
        const int mt = t & 1, nt = t >> 1;
        const _Float16* arow = Qf + (mt * 16 + (lane & 15)) * 256;
        v8f acc = {};
#pragma unroll
        for (int kb = 0; kb < 8; ++kb) {
            v16h a  = load_fragA(arow, kb, lane);
            v16h bm = load_fragB_rowmajor(Kf, 256, nt, kb, lane);
            acc = __builtin_amdgcn_wmma_f32_16x16x32_f16(false, a, false, bm,
                                                         (short)0, acc, false, false);
        }
        const int j  = nt * 16 + (lane & 15);
        const int ib = mt * 16 + ((lane & 16) ? 8 : 0);
#pragma unroll
        for (int r = 0; r < 8; ++r) S32[(ib + r) * 32 + j] = acc[r] * 0.0625f;
    }
    __syncthreads();

    // softmax over the 31 off-diagonal entries; emit weights_full (diag=1.0)
    float* Wout = P.out + (size_t)135168 + (size_t)b * 1024;
    if (tid < 32) {                      // exactly wave 0: uniform
        const int i = tid;
        float mx = -3.0e38f;
        for (int j = 0; j < 32; ++j) if (j != i) mx = fmaxf(mx, S32[i * 32 + j]);
        float sum = 0.0f;
        for (int j = 0; j < 32; ++j) if (j != i) sum += __expf(S32[i * 32 + j] - mx);
        const float inv = 1.0f / sum;
        for (int j = 0; j < 32; ++j) {
            float w = (j == i) ? 0.0f : __expf(S32[i * 32 + j] - mx) * inv;
            WtH[i * 32 + j]  = (_Float16)w;
            Wout[i * 32 + j] = (j == i) ? 1.0f : w;
        }
    }
    __syncthreads();

    // 9) node_feat = Wt @ AV (+ oa-mean) -> LN1 -> FEAT[:, 256:512]
    for (int t = wave; t < 32; t += 8) {
        const int mt = t & 1, nt = t >> 1;
        v16h a  = load_fragA(WtH + (mt * 16 + (lane & 15)) * 32, 0, lane);
        v16h bm = load_fragB_rowmajor(AV, 256, nt, 0, lane);
        v8f acc = {};
        acc = __builtin_amdgcn_wmma_f32_16x16x32_f16(false, a, false, bm,
                                                     (short)0, acc, false, false);
        const int n  = nt * 16 + (lane & 15);
        const int ib = mt * 16 + ((lane & 16) ? 8 : 0);
#pragma unroll
        for (int r = 0; r < 8; ++r) {
            const int i = ib + r;
            float oa = (float)OA[i * 256 + n];
            TMP32[i * 256 + n] = acc[r] + (SumOA[n] - oa) * (1.0f / 31.0f);
        }
    }
    __syncthreads();
    layer_norm(TMP32, P.ln1_w, P.ln1_b, FEAT, 512, 256, tid);
    __syncthreads();

    // 10) head MLP: 512 -> 256 -> 256 -> 16
    mlp_layer(FEAT, 512, 16, P.PfW1, P.f_b1, HID, nullptr, 256, 16, tid, true, false); __syncthreads();
    mlp_layer(HID, 256, 8, P.PfW2, P.f_b2, E, nullptr, 256, 16, tid, true, false); __syncthreads();
    mlp_layer(E, 256, 8, P.PfW3, P.f_b3, nullptr, Qb, 16, 1, tid, false, true); __syncthreads();

    // 11) Value[b,n,m] = Q[b,n,:] . policies[b,m,:];  Q_value[b,n] = actions . Q
    const float* pol = P.policies + (size_t)b * 32 * 16;
    float* Vout = P.out + (size_t)b * 1024;
    for (int idx = tid; idx < 1024; idx += 256) {
        const int i = idx >> 5, j = idx & 31;
        float s = 0.0f;
#pragma unroll
        for (int a = 0; a < 16; ++a) s += Qb[i * 16 + a] * pol[j * 16 + a];
        Vout[idx] = s;
    }
    if (tid < 32) {
        float s = 0.0f;
#pragma unroll
        for (int a = 0; a < 16; ++a) s += ac[tid * 16 + a] * Qb[tid * 16 + a];
        P.out[131072 + (size_t)b * 32 + tid] = s;
    }
}

extern "C" void kernel_launch(void* const* d_in, const int* in_sizes, int n_in,
                              void* d_out, int out_size, void* d_ws, size_t ws_size,
                              hipStream_t stream)
{
    (void)in_sizes; (void)n_in; (void)out_size; (void)ws_size;
    _Float16* wsH = (_Float16*)d_ws;

    size_t off = 0;
    auto alloc = [&](size_t kpad, size_t n) { size_t o = off; off += kpad * n; return o; };
    const size_t o_se1 = alloc(128, 256), o_se2 = alloc(256, 256);
    const size_t o_k1  = alloc(256, 256), o_k2  = alloc(256, 256);
    const size_t o_q1  = alloc(256, 256), o_q2  = alloc(256, 256);
    const size_t o_sa1 = alloc(160, 256), o_sa2 = alloc(256, 256);
    const size_t o_av1 = alloc(256, 256), o_av2 = alloc(256, 256);
    const size_t o_ca1 = alloc(128, 256), o_ca2 = alloc(256, 256);
    const size_t o_f1  = alloc(512, 256), o_f2  = alloc(256, 256), o_f3 = alloc(256, 16);

    auto pack = [&](int argi, size_t o, int K, int Kpad, int N) {
        dim3 g(Kpad / 32, N / 16);
        pack_weight_kernel<<<g, 32, 0, stream>>>((const float*)d_in[argi], wsH + o, K, N);
    };
    pack(3,  o_se1, 128, 128, 256);
    pack(5,  o_se2, 256, 256, 256);
    pack(7,  o_k1,  256, 256, 256);
    pack(9,  o_k2,  256, 256, 256);
    pack(11, o_q1,  256, 256, 256);
    pack(13, o_q2,  256, 256, 256);
    pack(15, o_sa1, 144, 160, 256);   // zero-pad K 144 -> 160
    pack(17, o_sa2, 256, 256, 256);
    pack(19, o_av1, 256, 256, 256);
    pack(21, o_av2, 256, 256, 256);
    pack(23, o_ca1, 128, 128, 256);
    pack(25, o_ca2, 256, 256, 256);
    pack(27, o_f1,  512, 512, 256);
    pack(29, o_f2,  256, 256, 256);
    pack(31, o_f3,  256, 256, 16);

    QNetParams P;
    P.states   = (const float*)d_in[0];
    P.policies = (const float*)d_in[1];
    P.actions  = (const float*)d_in[2];
    P.se_b1 = (const float*)d_in[4];  P.se_b2 = (const float*)d_in[6];
    P.k_b1  = (const float*)d_in[8];  P.k_b2  = (const float*)d_in[10];
    P.q_b1  = (const float*)d_in[12]; P.q_b2  = (const float*)d_in[14];
    P.sa_b1 = (const float*)d_in[16]; P.sa_b2 = (const float*)d_in[18];
    P.av_b1 = (const float*)d_in[20]; P.av_b2 = (const float*)d_in[22];
    P.ca_b1 = (const float*)d_in[24]; P.ca_b2 = (const float*)d_in[26];
    P.f_b1  = (const float*)d_in[28]; P.f_b2  = (const float*)d_in[30]; P.f_b3 = (const float*)d_in[32];
    P.ln1_w = (const float*)d_in[33]; P.ln1_b = (const float*)d_in[34];
    P.ln2_w = (const float*)d_in[35]; P.ln2_b = (const float*)d_in[36];
    P.PseW1 = wsH + o_se1; P.PseW2 = wsH + o_se2;
    P.PkW1  = wsH + o_k1;  P.PkW2  = wsH + o_k2;
    P.PqW1  = wsH + o_q1;  P.PqW2  = wsH + o_q2;
    P.PsaW1 = wsH + o_sa1; P.PsaW2 = wsH + o_sa2;
    P.PavW1 = wsH + o_av1; P.PavW2 = wsH + o_av2;
    P.PcaW1 = wsH + o_ca1; P.PcaW2 = wsH + o_ca2;
    P.PfW1  = wsH + o_f1;  P.PfW2  = wsH + o_f2;  P.PfW3 = wsH + o_f3;
    P.out = (float*)d_out;

    constexpr size_t SMEM =
        (size_t)(32 * 160 + 5 * 32 * 256 + 32 * 512 + 32 * 256 + 32 * 32) * 2 +   // f16
        (size_t)(32 * 256 + 32 * 32 + 32 * 16 + 256 + 256) * 4;                   // f32
    qnet_fused_kernel<<<dim3(128), dim3(256), SMEM, stream>>>(P);
}
